// AutoCorrelationMultiHead_35665408426442
// MI455X (gfx1250) — matrix operand order repaired and checked by reading, not compiled, not run on hardware
//
#include <hip/hip_runtime.h>
#include <hip/hip_bf16.h>

typedef __attribute__((ext_vector_type(16))) _Float16 v16h;
typedef __attribute__((ext_vector_type(8)))  _Float16 v8h;
typedef __attribute__((ext_vector_type(8)))  float    v8f;
typedef __attribute__((ext_vector_type(2)))  int      i32x2;
typedef __attribute__((ext_vector_type(4)))  int      i32x4;

#define BB   4
#define LL   4096
#define DMM  512
#define HH   8
#define DKK  64
#define KTOP 16
#define DCHUNK 4   // d-values per corr_fwd block -> 16 chunks per (b,h)

// ---- CDNA5 async global->LDS path (ASYNCcnt), guarded with fallback --------
#if defined(__has_builtin)
#if __has_builtin(__builtin_amdgcn_global_load_async_to_lds_b128) && \
    __has_builtin(__builtin_amdgcn_global_load_async_to_lds_b64)  && \
    __has_builtin(__builtin_amdgcn_s_wait_asynccnt)
#define HAS_ASYNC 1
#endif
#endif
#ifndef HAS_ASYNC
#define HAS_ASYNC 0
#endif

#if HAS_ASYNC
#define GPTR2(p) ((__attribute__((address_space(1))) i32x2*)(p))
#define LPTR2(p) ((__attribute__((address_space(3))) i32x2*)(p))
#define GPTR4(p) ((__attribute__((address_space(1))) i32x4*)(p))
#define LPTR4(p) ((__attribute__((address_space(3))) i32x4*)(p))
#endif

template <int N>
__device__ __forceinline__ void async_wait() {
#if HAS_ASYNC
  __builtin_amdgcn_s_wait_asynccnt(N);
#endif
}

// ---------------------------------------------------------------------------
// f32 -> f16 conversion
// ---------------------------------------------------------------------------
__global__ __launch_bounds__(256) void cvt_f32_to_f16(const float* __restrict__ in,
                                                      _Float16* __restrict__ out, int n) {
  int i = blockIdx.x * 256 + threadIdx.x;
  if (i < n) out[i] = (_Float16)in[i];
}

// ---------------------------------------------------------------------------
// WMMA GEMM: out = A(f16, Mx512 row-major) * W^T(f16, 512x512 row-major) + bias
// Block = 4 waves = 16x256 tile; K-steps staged through double-buffered LDS
// via global_load_async_to_lds (ASYNCcnt) so DMA of step i+1 overlaps WMMA of
// step i. One wave computes 16x64 via 4x v_wmma_f32_16x16x32_f16 per step.
// headLayout=1 stores (B,H,L,Dk); headLayout=0 stores (M,512) plain.
// ---------------------------------------------------------------------------
__device__ __forceinline__ void stage_copy(const _Float16* __restrict__ A,
                                           const _Float16* __restrict__ W,
                                           _Float16* aT, _Float16* bT,
                                           int m0, int nblk, int kk, int tid) {
  // A tile: 16 rows x 32 halfs (1KB): one b64 per thread (uniform per wave)
  {
    int e = tid * 4;                    // half index in tile
    int r = e >> 5, krel = e & 31;
    const _Float16* src = A + (size_t)(m0 + r) * DMM + kk + krel;
#if HAS_ASYNC
    __builtin_amdgcn_global_load_async_to_lds_b64(GPTR2(src), LPTR2(aT + e), 0, 0);
#else
    *(unsigned long long*)(aT + e) = *(const unsigned long long*)src;
#endif
  }
  // B tile: 256 cols x 32 halfs (16KB): 8 b128 per thread (uniform per wave)
#pragma unroll
  for (int j = 0; j < 8; ++j) {
    int idx = tid + j * 128;            // b128 unit, 1024 total
    int cn = idx >> 2, krel = (idx & 3) * 8;
    const _Float16* src = W + (size_t)(nblk + cn) * DMM + kk + krel;
#if HAS_ASYNC
    __builtin_amdgcn_global_load_async_to_lds_b128(GPTR4(src), LPTR4(bT + idx * 8), 0, 0);
#else
    *(uint4*)(bT + idx * 8) = *(const uint4*)src;
#endif
  }
}

__global__ __launch_bounds__(128) void gemm_wmma(const _Float16* __restrict__ A,
                                                 const _Float16* __restrict__ W,
                                                 const float* __restrict__ bias,
                                                 float* __restrict__ out,
                                                 int headLayout) {
  __shared__ __align__(16) _Float16 sA[2][16 * 32];     // 2 x 1KB
  __shared__ __align__(16) _Float16 sB[2][256 * 32];    // 2 x 16KB

  int tid   = threadIdx.x;
  int wid   = tid >> 5;
  int lane  = tid & 31;
  int mt    = blockIdx.x >> 1;            // 1024 m-tiles of 16
  int nblk  = (blockIdx.x & 1) * 256;     // block's 256-col slice
  int m0    = mt * 16;
  int wn    = wid * 64;                   // wave's block-local col base
  int lh    = lane >> 4;                  // lane half selects K sub-offsets
  int l15   = lane & 15;

  v8f acc[4] = {};

  stage_copy(A, W, sA[0], sB[0], m0, nblk, 0, tid);

  for (int kk = 0; kk < DMM; kk += 32) {
    int cur = (kk >> 5) & 1;
    bool hasNext = (kk + 32) < DMM;
    if (hasNext) stage_copy(A, W, sA[cur ^ 1], sB[cur ^ 1], m0, nblk, kk + 32, tid);

    // retire exactly the previous stage's 9 async ops; keep next in flight
    if (hasNext) async_wait<9>(); else async_wait<0>();
    __syncthreads();                      // stage `cur` visible to all waves

    const _Float16* aT = sA[cur];
    const _Float16* bT = sB[cur];

    // A-fragment: VGPR0-3 = K[lh*16 .. +7], VGPR4-7 = K[lh*16+8 .. +15]
    const _Float16* ap = aT + l15 * 32 + lh * 16;
    v8h alo = *(const v8h*)ap;
    v8h ahi = *(const v8h*)(ap + 8);
    v16h a;
#pragma unroll
    for (int i = 0; i < 8; ++i) { a[i] = alo[i]; a[8 + i] = ahi[i]; }

#pragma unroll
    for (int t = 0; t < 4; ++t) {
      // B-fragment: lane l15 = column, K-half = lh*16, 16 contiguous f16
      int cn = wn + t * 16 + l15;
      v16h b = *(const v16h*)(bT + cn * 32 + lh * 16);
      acc[t] = __builtin_amdgcn_wmma_f32_16x16x32_f16(false, a, false, b,
                                                      (short)0, acc[t], false, false);
    }
    __syncthreads();                      // all reads of `cur` done before reuse
  }

#pragma unroll
  for (int t = 0; t < 4; ++t) {
    int n = nblk + wn + t * 16 + l15;
    float bv = bias[n];
#pragma unroll
    for (int v = 0; v < 8; ++v) {
      int mrow = m0 + v + lh * 8;   // C/D layout: lanes0-15 M=v, lanes16-31 M=v+8
      float val = acc[t][v] + bv;
      if (headLayout) {
        int bidx = mrow >> 12, l = mrow & (LL - 1);
        int h = n >> 6, d = n & (DKK - 1);
        out[((size_t)((bidx * HH + h) << 12) + l) * DKK + d] = val;
      } else {
        out[(size_t)mrow * DMM + n] = val;
      }
    }
  }
}

// ---------------------------------------------------------------------------
// In-place radix-2 FFT of 4096 complex points in LDS (input bit-reversed).
// sign = -1 forward DFT, +1 inverse (unnormalized).
// ---------------------------------------------------------------------------
__device__ __forceinline__ void fft4096(float2* buf, int tid, float sign) {
  for (int s = 1; s <= 12; ++s) {
    int half = 1 << (s - 1);
    for (int idx = tid; idx < 2048; idx += 256) {
      int blk = idx >> (s - 1);
      int off = idx & (half - 1);
      int i0 = (blk << s) + off;
      int i1 = i0 + half;
      float ang = sign * 6.2831853071795864f * (float)off / (float)(1 << s);
      float sn, cs;
      __sincosf(ang, &sn, &cs);
      float2 u = buf[i0], v = buf[i1];
      float vr = cs * v.x - sn * v.y;
      float vi = cs * v.y + sn * v.x;
      buf[i0] = make_float2(u.x + vr, u.y + vi);
      buf[i1] = make_float2(u.x - vr, u.y - vi);
    }
    __syncthreads();
  }
}

// ---------------------------------------------------------------------------
// Forward stage: per (b,h,d-chunk) compute partial cross spectrum
// S_part[f] = sum_{d in chunk} Qf[f] * conj(Kf[f]) using 2-for-1 real FFT.
// Register accumulation -> deterministic (no atomics).
// ---------------------------------------------------------------------------
__global__ __launch_bounds__(256) void corr_fwd(const float* __restrict__ Q,
                                                const float* __restrict__ Kk,
                                                float2* __restrict__ Spart) {
  __shared__ float2 buf[LL];
  int bh    = blockIdx.x >> 4;   // 16 chunks per (b,h)
  int chunk = blockIdx.x & 15;
  int tid   = threadIdx.x;
  const float* qb = Q  + (size_t)bh * LL * DKK;
  const float* kb = Kk + (size_t)bh * LL * DKK;

  float sre[16] = {}, sim[16] = {};
  for (int dd = 0; dd < DCHUNK; ++dd) {
    int d = chunk * DCHUNK + dd;
    __syncthreads();
    for (int t = tid; t < LL; t += 256) {
      int r = __brev((unsigned)t) >> 20;          // 12-bit bit reversal
      buf[r] = make_float2(qb[(size_t)t * DKK + d], kb[(size_t)t * DKK + d]);
    }
    __syncthreads();
    fft4096(buf, tid, -1.0f);
#pragma unroll
    for (int j = 0; j < 16; ++j) {
      int f = tid + j * 256;
      float2 Z  = buf[f];
      float2 Zc = buf[(LL - f) & (LL - 1)];
      float qr = Z.x + Zc.x, qi = Z.y - Zc.y;     // 2*Qf
      float kr = Z.y + Zc.y, ki = Z.x - Zc.x;     // 2*conj(Kf) components
      sre[j] += qr * kr - qi * ki;
      sim[j] += qr * ki + kr * qi;
    }
  }
  const float scale = 0.25f / ((float)DKK * (float)LL); // 1/4 (trick) * 1/Dk * 1/N
#pragma unroll
  for (int j = 0; j < 16; ++j) {
    int f = tid + j * 256;
    Spart[((size_t)bh * 16 + chunk) * LL + f] = make_float2(sre[j] * scale, sim[j] * scale);
  }
}

// ---------------------------------------------------------------------------
// Inverse stage: sum partial spectra, inverse FFT -> corr_mean, top-16 +
// softmax. One block per (b,h).
// ---------------------------------------------------------------------------
__global__ __launch_bounds__(256) void corr_inv_topk(const float2* __restrict__ Spart,
                                                     int* __restrict__ delays,
                                                     float* __restrict__ wout) {
  __shared__ float2 buf[LL];
  __shared__ float redv[256];
  __shared__ int   redi[256];
  __shared__ float topv[KTOP];
  __shared__ int   topi[KTOP];
  int bh = blockIdx.x;
  int tid = threadIdx.x;

  for (int t = tid; t < LL; t += 256) {
    float sx = 0.f, sy = 0.f;
    for (int c = 0; c < 16; ++c) {
      float2 p = Spart[((size_t)bh * 16 + c) * LL + t];
      sx += p.x; sy += p.y;
    }
    buf[__brev((unsigned)t) >> 20] = make_float2(sx, sy);
  }
  __syncthreads();
  fft4096(buf, tid, +1.0f);   // Hermitian input -> real corr in .x

  for (int it = 0; it < KTOP; ++it) {
    float best = -3.4e38f; int bi = 1 << 30;
    for (int t = tid; t < LL; t += 256) {
      float v = buf[t].x;
      if (v > best || (v == best && t < bi)) { best = v; bi = t; }
    }
    redv[tid] = best; redi[tid] = bi;
    __syncthreads();
    for (int off = 128; off > 0; off >>= 1) {
      if (tid < off) {
        float v2 = redv[tid + off]; int i2 = redi[tid + off];
        if (v2 > redv[tid] || (v2 == redv[tid] && i2 < redi[tid])) {
          redv[tid] = v2; redi[tid] = i2;
        }
      }
      __syncthreads();
    }
    if (tid == 0) {
      topv[it] = redv[0]; topi[it] = redi[0];
      buf[redi[0]].x = -3.4e38f;  // mask winner
    }
    __syncthreads();
  }
  if (tid == 0) {
    float mx = topv[0];
    for (int i = 1; i < KTOP; ++i) mx = fmaxf(mx, topv[i]);
    float e[KTOP], sum = 0.f;
    for (int i = 0; i < KTOP; ++i) { e[i] = expf(topv[i] - mx); sum += e[i]; }
    for (int i = 0; i < KTOP; ++i) {
      wout[bh * KTOP + i]   = e[i] / sum;
      delays[bh * KTOP + i] = topi[i];
    }
  }
}

// ---------------------------------------------------------------------------
// Rolled gather: ctx[b,l,h*64+d] = sum_j w_j * V[b,h,(l-delay_j) mod L, d]
// writes f16 for the output-projection WMMA GEMM.
// ---------------------------------------------------------------------------
__global__ __launch_bounds__(256) void gather_ctx(const float* __restrict__ V,
                                                  const int* __restrict__ delays,
                                                  const float* __restrict__ w,
                                                  _Float16* __restrict__ ctx) {
  size_t gid = (size_t)blockIdx.x * 256 + threadIdx.x;  // B*H*L*Dk threads
  int d  = gid & (DKK - 1);
  int l  = (gid >> 6) & (LL - 1);
  int bh = (int)(gid >> 18);
  int bidx = bh >> 3, h = bh & 7;
  const float* vb = V + (size_t)bh * LL * DKK;
  float sum = 0.f;
#pragma unroll
  for (int j = 0; j < KTOP; ++j) {
    int   dl = delays[bh * KTOP + j];
    float wj = w[bh * KTOP + j];
    sum += wj * vb[((size_t)((l - dl) & (LL - 1)) << 6) + d];
  }
  ctx[((size_t)(bidx * LL + l)) * DMM + h * DKK + d] = (_Float16)sum;
}

// ---------------------------------------------------------------------------
extern "C" void kernel_launch(void* const* d_in, const int* in_sizes, int n_in,
                              void* d_out, int out_size, void* d_ws, size_t ws_size,
                              hipStream_t stream) {
  const float* x  = (const float*)d_in[0];
  const float* Wq = (const float*)d_in[1];
  const float* bq = (const float*)d_in[2];
  const float* Wk = (const float*)d_in[3];
  const float* bk = (const float*)d_in[4];
  const float* Wv = (const float*)d_in[5];
  const float* bv = (const float*)d_in[6];
  const float* Wo = (const float*)d_in[7];
  const float* bo = (const float*)d_in[8];
  float* out = (float*)d_out;

  const size_t M      = (size_t)BB * LL;          // 16384
  const size_t nX     = M * DMM;                  // 8388608
  const size_t nW     = (size_t)DMM * DMM;        // 262144
  const size_t BHn    = (size_t)BB * HH;          // 32

  char* ws = (char*)d_ws;
  size_t off = 0;
  auto carve = [&](size_t bytes) { char* p = ws + off; off += (bytes + 255) & ~(size_t)255; return p; };

  _Float16* x16  = (_Float16*)carve(nX * 2);      // 16 MB
  _Float16* wq16 = (_Float16*)carve(nW * 2);
  _Float16* wk16 = (_Float16*)carve(nW * 2);
  _Float16* wv16 = (_Float16*)carve(nW * 2);
  _Float16* wo16 = (_Float16*)carve(nW * 2);
  float*    Qh   = (float*)carve(nX * 4);         // 32 MB, (B,H,L,Dk)
  float*    Kh   = (float*)carve(nX * 4);
  float*    Vh   = (float*)carve(nX * 4);
  float2*   Sp   = (float2*)carve(BHn * 16 * LL * 8);  // 16 MB partial spectra
  int*      dly  = (int*)carve(BHn * KTOP * 4);
  float*    wsm  = (float*)carve(BHn * KTOP * 4);
  _Float16* ctx16 = (_Float16*)carve(nX * 2);     // 16 MB

  // 1) f16 copies of x and the four weight matrices
  cvt_f32_to_f16<<<(int)(nX / 256), 256, 0, stream>>>(x, x16, (int)nX);
  cvt_f32_to_f16<<<(int)(nW / 256), 256, 0, stream>>>(Wq, wq16, (int)nW);
  cvt_f32_to_f16<<<(int)(nW / 256), 256, 0, stream>>>(Wk, wk16, (int)nW);
  cvt_f32_to_f16<<<(int)(nW / 256), 256, 0, stream>>>(Wv, wv16, (int)nW);
  cvt_f32_to_f16<<<(int)(nW / 256), 256, 0, stream>>>(Wo, wo16, (int)nW);

  // 2) Q/K/V projections (WMMA + async-LDS double buffering), head layout
  const int gemmBlocks = (int)((M / 16) * 2);     // 2048
  gemm_wmma<<<gemmBlocks, 128, 0, stream>>>(x16, wq16, bq, Qh, 1);
  gemm_wmma<<<gemmBlocks, 128, 0, stream>>>(x16, wk16, bk, Kh, 1);
  gemm_wmma<<<gemmBlocks, 128, 0, stream>>>(x16, wv16, bv, Vh, 1);

  // 3) FFT cross-spectrum (deterministic register accumulation per chunk)
  corr_fwd<<<(int)(BHn * 16), 256, 0, stream>>>(Qh, Kh, Sp);

  // 4) inverse FFT + top-16 + softmax
  corr_inv_topk<<<(int)BHn, 256, 0, stream>>>(Sp, dly, wsm);

  // 5) rolled weighted gather of V -> ctx (f16)
  gather_ctx<<<(int)(nX / 256), 256, 0, stream>>>(Vh, dly, wsm, ctx16);

  // 6) output projection (WMMA), plain layout into d_out
  gemm_wmma<<<gemmBlocks, 128, 0, stream>>>(ctx16, wo16, bo, out, 0);
}
